// VQPTTokenizer_12309376271036
// MI455X (gfx1250) — compile-verified, matrix operand order checked by reading, not run on hardware
//
#include <hip/hip_runtime.h>
#include <hip/hip_bf16.h>
#include <math.h>

// ---------------------------------------------------------------------------
// CDNA5 (gfx1250) wave32 WMMA / TDM types
// ---------------------------------------------------------------------------
typedef _Float16     v16h  __attribute__((ext_vector_type(16)));
typedef float        v8f   __attribute__((ext_vector_type(8)));
typedef unsigned int u32x4 __attribute__((ext_vector_type(4)));
typedef int          i32x8 __attribute__((ext_vector_type(8)));
typedef int          i32x4 __attribute__((ext_vector_type(4)));

union AFrag { v16h v; _Float16 e[16]; };
union CFrag { v8f  v; float     e[8];  };

#if defined(__gfx1250__) && __has_builtin(__builtin_amdgcn_tensor_load_to_lds) && \
    __has_builtin(__builtin_amdgcn_s_wait_tensorcnt)
#define USE_TDM 1
#else
#define USE_TDM 0
#endif

#define TBM 32   // M tile per block
#define TBN 64   // N tile per block
#define TKT 64   // K staged per iteration (two WMMA sub-chunks of 32)

// ---------------------------------------------------------------------------
// Pack conv weights (OIHW f32, or (cin,cout,4,4) flipped for transposed conv)
// into K-major f16 panels wt[k][co], zero-padded to Kp x Np (64-multiples).
// ---------------------------------------------------------------------------
__global__ void pack_weights_kernel(const float* __restrict__ w,
                                    _Float16* __restrict__ wt,
                                    int Cin, int Cout, int KH, int KW,
                                    int transposed, int Kp, int Np)
{
    const int khw = KH * KW, Ktot = Cin * khw;
    long n = (long)Kp * Np;
    for (long i = (long)blockIdx.x * blockDim.x + threadIdx.x; i < n;
         i += (long)gridDim.x * blockDim.x) {
        int co = (int)(i & (Np - 1));     // Np is a power of two
        int k  = (int)(i / Np);
        float v = 0.f;
        if (k < Ktot && co < Cout) {
            int ci = k / khw, r = k - ci * khw;
            int kh = r / KW, kw = r - kh * KW;
            v = transposed
                    ? w[((size_t)(ci * Cout + co) * KH + (KH - 1 - kh)) * KW + (KW - 1 - kw)]
                    : w[((size_t)(co * Cin + ci) * KH + kh) * KW + kw];
        }
        wt[i] = (_Float16)v;
    }
}

// ---------------------------------------------------------------------------
// TDM: DMA one 64x64 f16 tile of the packed weight panel into LDS.
// D# per CDNA5 ISA ch.8 (6-arg builtin form).
// ---------------------------------------------------------------------------
#if USE_TDM
__device__ __forceinline__ void tdm_load_wtile(const _Float16* src,
                                               void* lds_dst, int Np)
{
    unsigned long long ga = (unsigned long long)(uintptr_t)src;
    unsigned ldsOff = (unsigned)(uintptr_t)lds_dst;   // low 32 bits = LDS byte addr
    u32x4 g0;
    g0[0] = 1u;                                        // count=1, user D#, no gather
    g0[1] = ldsOff;                                    // lds_addr
    g0[2] = (unsigned)(ga & 0xffffffffu);              // global_addr[31:0]
    g0[3] = (unsigned)((ga >> 32) & 0x1ffffffu) | (2u << 30); // addr[56:32], type=2
    unsigned td0 = (unsigned)Np;                       // tensor_dim0 (row length)
    unsigned long long s0 = (unsigned long long)Np;    // tensor_dim0_stride
    i32x8 g1;
    g1[0] = (int)(1u << 16);                           // wg_mask=0, data_size=1 (2B)
    g1[1] = (int)((td0 & 0xffffu) << 16);              // tensor_dim0[15:0] @bit48
    g1[2] = (int)((td0 >> 16) | (0xffffu << 16));      // td0[31:16] | tensor_dim1[15:0]
    g1[3] = (int)(0u | (64u << 16));                   // tensor_dim1[31:16]=0 | tile_dim0=64
    g1[4] = (int)(64u);                                // tile_dim1=64, tile_dim2=0
    g1[5] = (int)(s0 & 0xffffffffu);                   // stride0[31:0]
    g1[6] = (int)((s0 >> 32) & 0xffffu);               // stride0[47:32], stride1=0
    g1[7] = 0;
    i32x4 gz4; gz4[0] = gz4[1] = gz4[2] = gz4[3] = 0;
    i32x8 gz8;
#pragma unroll
    for (int i = 0; i < 8; ++i) gz8[i] = 0;
    __builtin_amdgcn_tensor_load_to_lds(g0, g1, gz4, gz4, gz8, 0);
}
#endif

// ---------------------------------------------------------------------------
// Implicit-GEMM conv, WMMA f32 <- f16 x f16.
// Block: 128 threads = 4 waves; tile 32(M) x 64(N); K staged 64/iter.
// All spatial extents are powers of two -> pure shift/mask addressing; the
// k -> (ci,kh,kw) decode is built once per K-chunk into a 64-entry LDS table.
// ---------------------------------------------------------------------------
__global__ __launch_bounds__(128)
void conv_wmma_kernel(const float* __restrict__ x, const _Float16* __restrict__ wt,
                      const float* __restrict__ bias, float* __restrict__ y,
                      int Cin, int Hin, int Win, int winSh,
                      int Cout, int Wout, int wSh, int hwSh,
                      int KH, int KW, int stride, int pad,
                      int transposed, int Cy_total, int co_off, int Np, int M)
{
    __shared__ _Float16 As[TBM][TKT];   // 4 KB
    __shared__ _Float16 Bs[TKT][TBN];   // 8 KB
    __shared__ int s_coff[TKT];         // k -> ci * Hin*Win  (-1 = out of range)
    __shared__ int s_kh[TKT];
    __shared__ int s_kw[TKT];

    const int tid  = threadIdx.x;
    const int lane = tid & 31;
    const int wid  = tid >> 5;           // 0..3
    const int wm   = wid >> 1;           // 0..1  M sub-tile
    const int np   = wid & 1;            // 0..1  N pair
    const int hi8  = (lane >= 16) ? 8 : 0;
    const int l15  = lane & 15;

    const int khw  = KH * KW;
    const int Ktot = Cin * khw;
    const int m0   = blockIdx.x * TBM;
    const int n0   = blockIdx.y * TBN;

    // HoWo = 1<<hwSh is a multiple of 32 and m0 is 32-aligned: the whole tile
    // lives in one image.
    const int nimg = m0 >> hwSh;
    const float* xbase = x + ((size_t)nimg * Cin << (hwSh > 0 ? 0 : 0)) * 0 +
                         (size_t)nimg * Cin * ((size_t)Hin << winSh);

    // per-thread fixed A-gather state: thread owns row mi, K-quarter q
    const int mi = tid >> 2;             // 0..31
    const int q  = tid & 3;              // 0..3  (kk = q*16 + j)
    const int m  = m0 + mi;
    const bool mok = (m < M);
    const int sp = m - (nimg << hwSh);
    const int oh = sp >> wSh;
    const int ow = sp & (Wout - 1);

    CFrag acc0, acc1;
#pragma unroll
    for (int i = 0; i < 8; ++i) { acc0.e[i] = 0.f; acc1.e[i] = 0.f; }

    for (int k0 = 0; k0 < Ktot; k0 += TKT) {
        // ---- B tile: 64x64 f16 panel slice ----------------------------------
#if USE_TDM
        if (wid == 0)
            tdm_load_wtile(wt + (size_t)k0 * Np + n0, &Bs[0][0], Np);
#else
        for (int i = tid; i < TKT * TBN; i += 128) {
            int kk = i >> 6, ni = i & 63;
            Bs[kk][ni] = wt[(size_t)(k0 + kk) * Np + n0 + ni];
        }
#endif
        // ---- per-chunk k decode table ---------------------------------------
        if (tid < TKT) {
            int k = k0 + tid;
            int coff = -1, kh = 0, kw = 0;
            if (k < Ktot) {
                int ci = k / khw, r = k - ci * khw;
                kh = r / KW; kw = r - kh * KW;
                coff = ci << (winSh) ;                 // ci * Win ...
                coff = ci * (Hin << winSh);            // ci * Hin * Win
            }
            s_coff[tid] = coff; s_kh[tid] = kh; s_kw[tid] = kw;
        }
        __syncthreads();

        // ---- A tile: im2col gather f32 -> f16 -------------------------------
        __builtin_prefetch(xbase + (size_t)((k0 + TKT) / khw) * (Hin << winSh), 0, 1);
#pragma unroll 4
        for (int j = 0; j < 16; ++j) {
            int kk = q * 16 + j;
            int coff = s_coff[kk];
            float v = 0.f;
            if (mok && coff >= 0) {
                int kh = s_kh[kk], kw = s_kw[kk];
                int ih, iw; bool ok;
                if (!transposed) {
                    ih = oh * stride - pad + kh;
                    iw = ow * stride - pad + kw;
                    ok = (ih >= 0 && ih < Hin && iw >= 0 && iw < Win);
                } else {            // ConvTranspose2d(4,2,1): dilated-input gather
                    int th = oh + kh - 2, tw = ow + kw - 2;
                    ok = (th >= 0 && tw >= 0 && !(th & 1) && !(tw & 1));
                    ih = th >> 1; iw = tw >> 1;
                    ok = ok && (ih < Hin) && (iw < Win);
                }
                if (ok) v = xbase[coff + (ih << winSh) + iw];
            }
            As[mi][kk] = (_Float16)v;
        }
#if USE_TDM
        if (wid == 0) __builtin_amdgcn_s_wait_tensorcnt((short)0);
#endif
        __syncthreads();

        // ---- 2 K sub-chunks x 2 N sub-tiles = 4 WMMAs -----------------------
#pragma unroll
        for (int kc = 0; kc < TKT; kc += 32) {
            AFrag af, bf0, bf1;
#pragma unroll
            for (int v2 = 0; v2 < 8; ++v2) {
                int kb = ((v2 & 3) << 1) + ((v2 >> 2) << 4) + hi8;
                af.e[2 * v2]      = As[wm * 16 + l15][kc + kb];
                af.e[2 * v2 + 1]  = As[wm * 16 + l15][kc + kb + 1];
                bf0.e[2 * v2]     = Bs[kc + kb][np * 32 + l15];
                bf0.e[2 * v2 + 1] = Bs[kc + kb + 1][np * 32 + l15];
                bf1.e[2 * v2]     = Bs[kc + kb][np * 32 + 16 + l15];
                bf1.e[2 * v2 + 1] = Bs[kc + kb + 1][np * 32 + 16 + l15];
            }
            acc0.v = __builtin_amdgcn_wmma_f32_16x16x32_f16(
                false, af.v, false, bf0.v, (short)0, acc0.v, false, false);
            acc1.v = __builtin_amdgcn_wmma_f32_16x16x32_f16(
                false, af.v, false, bf1.v, (short)0, acc1.v, false, false);
        }
        __syncthreads();
    }

    // ---- write out: C/D layout VGPR r -> M = r + hi8, N = lane&15 ----------
    const size_t ybase = ((size_t)nimg * Cy_total + co_off) << hwSh;
#pragma unroll
    for (int t = 0; t < 2; ++t) {
        CFrag* a = t ? &acc1 : &acc0;
#pragma unroll
        for (int r = 0; r < 8; ++r) {
            int mm = m0 + wm * 16 + r + hi8;
            int co = n0 + np * 32 + t * 16 + l15;
            if (mm < M && co < Cout) {
                int spn = mm - (nimg << hwSh);
                y[ybase + ((size_t)co << hwSh) + spn] = a->e[r] + bias[co];
            }
        }
    }
}

// ---------------------------------------------------------------------------
// GroupNorm stats per (image, group): 32 groups, contiguous channels
// ---------------------------------------------------------------------------
__global__ __launch_bounds__(256)
void gn_stats_kernel(const float* __restrict__ x, float* __restrict__ stats,
                     int C, int HW)
{
    int n = blockIdx.x >> 5, g = blockIdx.x & 31;
    int Cg = C / 32;
    const float* base = x + ((size_t)n * C + (size_t)g * Cg) * HW;
    long cnt = (long)Cg * HW;
    float s = 0.f, s2 = 0.f;
    for (long i = threadIdx.x; i < cnt; i += 256) {
        float v = base[i]; s += v; s2 += v * v;
    }
    __shared__ float sh[256], sh2[256];
    sh[threadIdx.x] = s; sh2[threadIdx.x] = s2;
    __syncthreads();
    for (int st = 128; st > 0; st >>= 1) {
        if (threadIdx.x < st) {
            sh[threadIdx.x]  += sh[threadIdx.x + st];
            sh2[threadIdx.x] += sh2[threadIdx.x + st];
        }
        __syncthreads();
    }
    if (threadIdx.x == 0) {
        float inv = 1.f / (float)cnt;
        float mu  = sh[0] * inv;
        float var = sh2[0] * inv - mu * mu;
        stats[(n * 32 + g) * 2]     = mu;
        stats[(n * 32 + g) * 2 + 1] = rsqrtf(var + 1e-5f);
    }
}

// GN affine + GEGLU (exact gelu). HW, Chalf, Cg are powers of two.
__global__ void gn_geglu_kernel(const float* __restrict__ t,
                                const float* __restrict__ stats,
                                const float* __restrict__ gamma,
                                const float* __restrict__ beta,
                                float* __restrict__ out,
                                int NB, int Chalf, int chSh, int HW, int hwSh,
                                int cgSh)
{
    int C = 2 * Chalf;
    long total = (long)NB << (chSh + hwSh);
    for (long i = (long)blockIdx.x * blockDim.x + threadIdx.x; i < total;
         i += (long)gridDim.x * blockDim.x) {
        int hw = (int)(i & (HW - 1));
        long r = i >> hwSh;
        int c = (int)(r & (Chalf - 1));
        int n = (int)(r >> chSh);
        int ca = c, cg = c + Chalf;
        float va = t[(((size_t)n * C + ca) << hwSh) + hw];
        float vg = t[(((size_t)n * C + cg) << hwSh) + hw];
        int gA = ca >> cgSh, gG = cg >> cgSh;
        float a  = (va - stats[(n * 32 + gA) * 2]) * stats[(n * 32 + gA) * 2 + 1] * gamma[ca] + beta[ca];
        float gt = (vg - stats[(n * 32 + gG) * 2]) * stats[(n * 32 + gG) * 2 + 1] * gamma[cg] + beta[cg];
        float gelu = 0.5f * a * (1.f + erff(a * 0.70710678118654752f));
        out[i] = gt * gelu;
    }
}

__global__ void add_kernel(const float* a, const float* b, float* o, long n) {
    for (long i = (long)blockIdx.x * blockDim.x + threadIdx.x; i < n;
         i += (long)gridDim.x * blockDim.x)
        o[i] = a[i] + b[i];
}

// copy src (NB,Cs,HW) into dst (NB,Cd,HW) at channel offset coff; pow2 dims
__global__ void copychan_kernel(const float* src, float* dst,
                                int NB, int Cs, int csSh, int HW, int hwSh,
                                int Cd, int coff) {
    long n = (long)NB << (csSh + hwSh);
    for (long i = (long)blockIdx.x * blockDim.x + threadIdx.x; i < n;
         i += (long)gridDim.x * blockDim.x) {
        int hw = (int)(i & (HW - 1));
        long r = i >> hwSh;
        int c  = (int)(r & (Cs - 1));
        int nn = (int)(r >> csSh);
        dst[(((size_t)nn * Cd + coff + c) << hwSh) + hw] = src[i];
    }
}

__global__ void csq_kernel(const float* cb, float* csq) {
    int c = blockIdx.x * blockDim.x + threadIdx.x;
    if (c < 1024) {
        float s = 0.f;
        for (int d = 0; d < 64; ++d) { float v = cb[c * 64 + d]; s += v * v; }
        csq[c] = s;
    }
}

// ---------------------------------------------------------------------------
// Multi-head VQ: one wave per (image*head, 16-position tile).
// ---------------------------------------------------------------------------
__global__ __launch_bounds__(32)
void vq_kernel(const float* __restrict__ z, const float* __restrict__ cb,
               const float* __restrict__ csq, float* __restrict__ quant,
               int* __restrict__ ids, float* __restrict__ lossacc)
{
    const int bh = blockIdx.x;
    const int b = bh >> 2, h = bh & 3;
    const int m0 = blockIdx.y * 16;
    const int lane = threadIdx.x;
    const int hi8 = (lane >= 16) ? 8 : 0;
    const int l15 = lane & 15;

    __shared__ float    zfs[16][64];
    __shared__ _Float16 cbs[64][16];
    __shared__ float    zsq[16];
    __shared__ float    mind[16];
    __shared__ int      mini[16];

    for (int i = lane; i < 16 * 64; i += 32) {
        int mm = i >> 6, d = i & 63;
        zfs[mm][d] = z[((size_t)b * 256 + h * 64 + d) * 1024 + (m0 + mm)];
    }
    __syncthreads();
    if (lane < 16) {
        float s = 0.f;
        for (int d = 0; d < 64; ++d) { float v = zfs[lane][d]; s += v * v; }
        zsq[lane] = s; mind[lane] = 3.4e38f; mini[lane] = 0;
    }
    __syncthreads();

    for (int t = 0; t < 64; ++t) {
        const int c0 = t * 16;
        for (int i = lane; i < 64 * 16; i += 32) {
            int d = i >> 4, j = i & 15;
            cbs[d][j] = (_Float16)cb[(size_t)(c0 + j) * 64 + d];
        }
        __syncthreads();

        CFrag acc;
#pragma unroll
        for (int i = 0; i < 8; ++i) acc.e[i] = 0.f;
#pragma unroll
        for (int kc = 0; kc < 64; kc += 32) {
            AFrag af, bf;
#pragma unroll
            for (int v2 = 0; v2 < 8; ++v2) {
                int kb = ((v2 & 3) << 1) + ((v2 >> 2) << 4) + hi8;
                af.e[2 * v2]     = (_Float16)zfs[l15][kc + kb];
                af.e[2 * v2 + 1] = (_Float16)zfs[l15][kc + kb + 1];
                bf.e[2 * v2]     = cbs[kc + kb][l15];
                bf.e[2 * v2 + 1] = cbs[kc + kb + 1][l15];
            }
            acc.v = __builtin_amdgcn_wmma_f32_16x16x32_f16(
                false, af.v, false, bf.v, (short)0, acc.v, false, false);
        }
#pragma unroll
        for (int r = 0; r < 8; ++r) {
            int mm = r + hi8;
            int code = c0 + l15;
            float bv = zsq[mm] - 2.f * acc.e[r] + csq[code];
            int bi = code;
#pragma unroll
            for (int s = 8; s >= 1; s >>= 1) {
                float ov = __shfl_xor(bv, s, 32);
                int   oi = __shfl_xor(bi, s, 32);
                if (ov < bv || (ov == bv && oi < bi)) { bv = ov; bi = oi; }
            }
            if (l15 == 0) {
                if (bv < mind[mm] || (bv == mind[mm] && bi < mini[mm])) {
                    mind[mm] = bv; mini[mm] = bi;
                }
            }
        }
        __syncthreads();
    }

    if (lane < 16) ids[(size_t)bh * 1024 + m0 + lane] = mini[lane];
    float ls = 0.f;
    for (int i = lane; i < 16 * 64; i += 32) {
        int mm = i >> 6, d = i & 63;
        float qv = cb[(size_t)mini[mm] * 64 + d];
        quant[((size_t)b * 256 + h * 64 + d) * 1024 + (m0 + mm)] = qv;
        float df = qv - zfs[mm][d];
        ls += df * df;
    }
#pragma unroll
    for (int s = 16; s >= 1; s >>= 1) ls += __shfl_xor(ls, s, 32);
    if (lane == 0) atomicAdd(lossacc, ls);
}

__global__ void tanh_kernel(const float* a, float* o, long n) {
    for (long i = (long)blockIdx.x * blockDim.x + threadIdx.x; i < n;
         i += (long)gridDim.x * blockDim.x)
        o[i] = tanhf(a[i]);
}

__global__ void finalize_kernel(const int* ids, const float* lossacc,
                                float* out_ids, float* out_loss) {
    long n = 32768;
    for (long i = (long)blockIdx.x * blockDim.x + threadIdx.x; i < n;
         i += (long)gridDim.x * blockDim.x)
        out_ids[i] = (float)ids[i];
    if (blockIdx.x == 0 && threadIdx.x == 0)
        out_loss[0] = 0.25f * lossacc[0] / (float)(8 * 4 * 1024 * 64);
}

// ---------------------------------------------------------------------------
// Host-side orchestration
// ---------------------------------------------------------------------------
static inline int ilog2i(int v) { int s = 0; while ((1 << s) < v) ++s; return s; }

extern "C" void kernel_launch(void* const* d_in, const int* in_sizes, int n_in,
                              void* d_out, int out_size, void* d_ws, size_t ws_size,
                              hipStream_t stream)
{
    (void)out_size; (void)ws_size;
    const int NB = 8;
    int pbase, vidx;
    if (in_sizes[0] == 2 * 4 * 3 * 128 * 128) { vidx = 0; pbase = 1; }
    else                                      { vidx = n_in - 1; pbase = 0; }
    auto P = [&](int i) -> const float* { return (const float*)d_in[pbase + i]; };
    const float* videos = (const float*)d_in[vidx];
    float* out = (float*)d_out;

    // ---- workspace carve (float units)
    float* ws = (float*)d_ws;
    size_t o = 0;
    float* S0 = ws + o; o += (size_t)NB * 64 * 128 * 128;
    float* S1 = ws + o; o += (size_t)NB * 64 * 64 * 64;
    float* Z  = ws + o; o += (size_t)NB * 256 * 32 * 32;
    float* Q  = ws + o; o += (size_t)NB * 256 * 32 * 32;
    float* A  = ws + o; o += (size_t)NB * 128 * 128 * 128;
    float* Bb = ws + o; o += (size_t)NB * 64 * 128 * 128;
    float* Cc = ws + o; o += (size_t)NB * 64 * 128 * 128;
    float* D  = ws + o; o += (size_t)NB * 128 * 128 * 128;
    float* STATS = ws + o; o += 512;
    float* CSQ   = ws + o; o += 1024;
    int*   IDS   = (int*)(ws + o); o += 32768;
    float* LACC  = ws + o; o += 1;
    _Float16* WT = (_Float16*)(ws + o); o += 3 * 1024 * 1024; // 6M f16 capacity

    (void)hipMemsetAsync(LACC, 0, sizeof(float), stream);

    size_t wtoff = 0;
    auto packw = [&](int wi, int Cin, int Cout, int KH, int transposed,
                     int& Np) -> const _Float16* {
        int Ktot = Cin * KH * KH;
        int Kp = (Ktot + TKT - 1) & ~(TKT - 1);
        Np = (Cout + 63) & ~63;
        _Float16* dst = WT + wtoff;
        wtoff += (size_t)Kp * Np;
        long n = (long)Kp * Np;
        pack_weights_kernel<<<(int)((n + 255) / 256), 256, 0, stream>>>(
            P(wi), dst, Cin, Cout, KH, KH, transposed, Kp, Np);
        return dst;
    };
    auto conv = [&](const float* x, int wi, int bi, float* y,
                    int Cin, int Hin, int Cout, int KH, int stride, int pad,
                    int transposed, int Cy_total, int co_off) {
        int Np;
        const _Float16* wt = packw(wi, Cin, Cout, KH, transposed, Np);
        int Hout = transposed ? Hin * 2 : (Hin + 2 * pad - KH) / stride + 1;
        int M = NB * Hout * Hout;
        int wSh = ilog2i(Hout), hwSh = 2 * wSh, winSh = ilog2i(Hin);
        dim3 grid((M + TBM - 1) / TBM, (Cout + TBN - 1) / TBN);
        conv_wmma_kernel<<<grid, 128, 0, stream>>>(
            x, wt, P(bi), y, Cin, Hin, Hin, winSh,
            Cout, Hout, wSh, hwSh, KH, KH, stride, pad,
            transposed, Cy_total, co_off, Np, M);
    };
    auto gn_geglu = [&](const float* t, int gi, int bti, float* outp,
                        int Chalf, int HW) {
        gn_stats_kernel<<<NB * 32, 256, 0, stream>>>(t, STATS, 2 * Chalf, HW);
        long total = (long)NB * Chalf * HW;
        int cgSh = ilog2i((2 * Chalf) / 32);
        gn_geglu_kernel<<<(int)((total + 255) / 256), 256, 0, stream>>>(
            t, STATS, P(gi), P(bti), outp, NB, Chalf, ilog2i(Chalf),
            HW, ilog2i(HW), cgSh);
    };
    auto addf = [&](const float* a, const float* b, float* op, long n) {
        add_kernel<<<(int)((n + 255) / 256), 256, 0, stream>>>(a, b, op, n);
    };
    auto resblock = [&](const float* X, float* G, float* SK, float* OUT,
                        int cin, int cout, int HWdim,
                        int c1w, int c1b, int c1g, int c1bt,
                        int c2w, int c2b, int c2g, int c2bt,
                        int skw, int skb) {
        int HW = HWdim * HWdim;
        conv(X, c1w, c1b, A, cin, HWdim, 2 * cout, 3, 1, 1, 0, 2 * cout, 0);
        gn_geglu(A, c1g, c1bt, G, cout, HW);
        conv(G, c2w, c2b, A, cout, HWdim, 2 * cout, 3, 1, 1, 0, 2 * cout, 0);
        gn_geglu(A, c2g, c2bt, G, cout, HW);
        long n = (long)NB * cout * HW;
        if (skw >= 0) {
            conv(X, skw, skb, SK, cin, HWdim, cout, 1, 1, 0, 0, cout, 0);
            addf(G, SK, OUT, n);
        } else {
            addf(G, X, OUT, n);
        }
    };
    auto copychan = [&](const float* src, float* dst, int Cs, int HW, int Cd, int coff) {
        long n = (long)NB * Cs * HW;
        copychan_kernel<<<(int)((n + 255) / 256), 256, 0, stream>>>(
            src, dst, NB, Cs, ilog2i(Cs), HW, ilog2i(HW), Cd, coff);
    };

    // ================= encoder =================
    resblock(videos, Bb, Cc, S0, 3, 64, 128, 37, 36, 38, 35, 41, 40, 42, 39, 44, 43);
    conv(S0, 4, 3, Bb, 64, 128, 64, 3, 1, 1, 0, 64, 0);
    conv(Bb, 2, 1, Cc, 64, 128, 64, 4, 2, 1, 0, 64, 0);
    resblock(Cc, Bb, nullptr, S1, 64, 64, 64, 7, 6, 8, 5, 11, 10, 12, 9, -1, -1);
    conv(S1, 16, 15, Bb, 64, 64, 64, 3, 1, 1, 0, 64, 0);
    conv(Bb, 14, 13, Cc, 64, 64, 128, 4, 2, 1, 0, 128, 0);
    resblock(Cc, Bb, nullptr, D, 128, 128, 32, 19, 18, 20, 17, 23, 22, 24, 21, -1, -1);
    conv(D, 46, 45, Z, 128, 32, 256, 3, 1, 1, 0, 256, 0);

    // ================= VQ =================
    csq_kernel<<<4, 256, 0, stream>>>(P(0), CSQ);
    {
        dim3 grid(NB * 4, 64);
        vq_kernel<<<grid, 32, 0, stream>>>(Z, P(0), CSQ, Q, IDS, LACC);
    }

    // ================= decoder =================
    conv(Q, 58, 57, D, 256, 32, 64, 4, 2, 1, 1, 128, 0);
    copychan(S1, D, 64, 64 * 64, 128, 64);
    resblock(D, Bb, Cc, Bb, 128, 64, 64, 49, 48, 50, 47, 53, 52, 54, 51, 56, 55);
    conv(Bb, 70, 69, D, 64, 64, 64, 4, 2, 1, 1, 128, 0);
    copychan(S0, D, 64, 128 * 128, 128, 64);
    resblock(D, Bb, Cc, Bb, 128, 64, 128, 61, 60, 62, 59, 65, 64, 66, 63, 68, 67);
    resblock(Bb, Cc, nullptr, Cc, 64, 64, 128, 29, 28, 30, 27, 33, 32, 34, 31, -1, -1);
    conv(Cc, 26, 25, A, 64, 128, 3, 3, 1, 1, 0, 3, 0);
    {
        long n = (long)NB * 3 * 128 * 128;
        tanh_kernel<<<(int)((n + 255) / 256), 256, 0, stream>>>(A, out, n);
    }
    finalize_kernel<<<128, 256, 0, stream>>>(IDS, LACC, out + 393216, out + 425984);
}
